// Attention_4844723110037
// MI455X (gfx1250) — compile-verified, matrix operand order checked by reading, not run on hardware
//
#include <hip/hip_runtime.h>
#include <hip/hip_bf16.h>

typedef __attribute__((ext_vector_type(16))) _Float16 v16h;
typedef __attribute__((ext_vector_type(8)))  _Float16 v8h;
typedef __attribute__((ext_vector_type(8)))  float    v8f;
typedef int v4i __attribute__((vector_size(16)));

#define D_DIM 512
#define H_DIM 512
#define S_LEN 4096
#define B_SZ  4

#if defined(__has_builtin)
#if __has_builtin(__builtin_amdgcn_global_load_async_to_lds_b128) && \
    __has_builtin(__builtin_amdgcn_s_wait_asynccnt)
#define USE_ASYNC_LDS 1
#endif
#endif
#ifndef USE_ASYNC_LDS
#define USE_ASYNC_LDS 0
#endif

#if USE_ASYNC_LDS
typedef __attribute__((address_space(1))) v4i* gptr_b128;  // global src
typedef __attribute__((address_space(3))) v4i* lptr_b128;  // LDS dst
#endif

__device__ __forceinline__ v8f wmma_f16(v16h a, v16h b, v8f c) {
  // D = A(16x32 f16) * B(32x16 f16) + C(16x16 f32)
  return __builtin_amdgcn_wmma_f32_16x16x32_f16(false, a, false, b, (short)0, c,
                                                false, false);
}

// Build an A/B operand fragment from 2 aligned 16B loads (no per-element VALU).
// Per-lane pattern: elems 0..7 at p[0..7], elems 8..15 at p[16..23].
__device__ __forceinline__ v16h load_frag(const _Float16* p) {
  v8h lo = *(const v8h*)(p);
  v8h hi = *(const v8h*)(p + 16);
  return __builtin_shufflevector(lo, hi, 0, 1, 2, 3, 4, 5, 6, 7, 8, 9, 10, 11,
                                 12, 13, 14, 15);
}

// ---------------------------------------------------------------------------
// Weights -> f16, transposed so every WMMA B-operand load is contiguous.
// Wq/Wk/Wv are [D,H] -> W*T[h][d];   Wo is [H,D] -> WoT[d][h]
// ---------------------------------------------------------------------------
__global__ void prep_weights_kernel(const float* __restrict__ Wq,
                                    const float* __restrict__ Wk,
                                    const float* __restrict__ Wv,
                                    const float* __restrict__ Wo,
                                    _Float16* __restrict__ WqT,
                                    _Float16* __restrict__ WkT,
                                    _Float16* __restrict__ WvT,
                                    _Float16* __restrict__ WoT) {
  int t = blockIdx.x * blockDim.x + threadIdx.x;  // 0 .. 512*512-1
  int r = t / 512, c = t % 512;
  WqT[c * 512 + r] = (_Float16)Wq[t];
  WkT[c * 512 + r] = (_Float16)Wk[t];
  WvT[c * 512 + r] = (_Float16)Wv[t];
  WoT[c * 512 + r] = (_Float16)Wo[t];
}

// ---------------------------------------------------------------------------
// Projection: Out = f16( X(f32)[R,D] * W[D,H] + bias ), W given as WT[H][D].
// One wave computes a 64x16 tile (4 row-tiles share each B fragment).
// transpose_out=1 -> Out laid out [B][H][S] (used for V).
// ---------------------------------------------------------------------------
__global__ __launch_bounds__(32) void proj_kernel(
    const float* __restrict__ X, const _Float16* __restrict__ WT,
    const float* __restrict__ bias, _Float16* __restrict__ Out,
    int transpose_out) {
  const int l = threadIdx.x;
  const int half = l >> 4, idx = l & 15;
  const int col0 = blockIdx.x * 16;
  const int row0 = blockIdx.y * 64;

  v8f acc[4];
#pragma unroll
  for (int rt = 0; rt < 4; ++rt) acc[rt] = (v8f){};

  const float*    ar = X  + (size_t)(row0 + idx) * D_DIM + half * 8;
  const _Float16* br = WT + (size_t)(col0 + idx) * D_DIM + half * 8;
  for (int k0 = 0; k0 < D_DIM; k0 += 32) {
    const v16h bf = load_frag(br + k0);
#pragma unroll
    for (int rt = 0; rt < 4; ++rt) {
      const float* arp = ar + (size_t)rt * 16 * D_DIM + k0;
      v16h a;
#pragma unroll
      for (int j = 0; j < 8; ++j) {
        a[j]     = (_Float16)arp[j];
        a[8 + j] = (_Float16)arp[16 + j];
      }
      acc[rt] = wmma_f16(a, bf, acc[rt]);
    }
  }
  const float bvl = bias[col0 + idx];
#pragma unroll
  for (int rt = 0; rt < 4; ++rt) {
    const int rbase = row0 + rt * 16;
    if (!transpose_out) {
#pragma unroll
      for (int r = 0; r < 8; ++r)
        Out[(size_t)(rbase + r + 8 * half) * H_DIM + col0 + idx] =
            (_Float16)(acc[rt][r] + bvl);
    } else {
      const int bb = rbase / S_LEN, sr = rbase % S_LEN;
      _Float16* op = Out + (size_t)bb * H_DIM * S_LEN +
                     (size_t)(col0 + idx) * S_LEN + sr + 8 * half;
#pragma unroll
      for (int r = 0; r < 8; ++r) op[r] = (_Float16)(acc[rt][r] + bvl);
    }
  }
}

// ---------------------------------------------------------------------------
// Flash attention, cooperative block: 4 waves, one 16-query tile.
// Per 128-key iteration: wave w computes the score tile for keys
// [kc+32w, kc+32w+32) (no redundancy), block-combines softmax stats via LDS,
// then each wave does P(16x128) x V(128x16) over its 128-wide H slice.
// Barrier scheme (2 per iter): barrier1 publishes smax, barrier2 publishes
// Pt/ssum; barrier1 of the next iteration already orders Pt reuse.
// ---------------------------------------------------------------------------
__global__ __launch_bounds__(128) void attn_kernel(
    const _Float16* __restrict__ Qh,  // [B*S, H]
    const _Float16* __restrict__ Kh,  // [B*S, H]
    const _Float16* __restrict__ Vt,  // [B][H][S]
    _Float16* __restrict__ Yh) {      // [B*S, H]
  __shared__ __align__(16) _Float16 Qs[16 * H_DIM];  // 16 KB query tile
  __shared__ __align__(16) _Float16 Pt[128][16];     // P^T [key][qrow], 4 KB
  __shared__ float smax[4][16];
  __shared__ float ssum[4][16];

  const int bb   = blockIdx.y;
  const int q0   = blockIdx.x * 16;
  const int wave = threadIdx.x >> 5;
  const int l    = threadIdx.x & 31;
  const int half = l >> 4, idx = l & 15;
  const int h0   = wave * 128;

  // ---- stage the (contiguous) 16x512 query tile into LDS
  const _Float16* Qg = Qh + (size_t)(bb * S_LEN + q0) * H_DIM;
#if USE_ASYNC_LDS
  for (int t = threadIdx.x; t < 16 * H_DIM / 8; t += 128)
    __builtin_amdgcn_global_load_async_to_lds_b128(
        (gptr_b128)(Qg + (size_t)t * 8), (lptr_b128)(Qs + (size_t)t * 8), 0, 0);
  __builtin_amdgcn_s_wait_asynccnt(0);
#else
  for (int t = threadIdx.x; t < 16 * H_DIM / 8; t += 128)
    ((v8h*)Qs)[t] = ((const v8h*)Qg)[t];
#endif
  __syncthreads();

  v8f zero = {};
  v8f acc[8];
#pragma unroll
  for (int t = 0; t < 8; ++t) acc[t] = zero;
  float mrow[8], srow[8];
#pragma unroll
  for (int r = 0; r < 8; ++r) { mrow[r] = -3.0e38f; srow[r] = 0.0f; }

  for (int kc = 0; kc < S_LEN; kc += 128) {
    const int myk = kc + wave * 32;
    // ---- scores for this wave's 32-key chunk (two 16x16 f32 tiles)
    v8f s0 = zero, s1 = zero;
    const _Float16* k0p =
        Kh + (size_t)(bb * S_LEN + myk + idx) * H_DIM + half * 8;
    const _Float16* k1p = k0p + (size_t)16 * H_DIM;
    const _Float16* qp  = Qs + idx * H_DIM + half * 8;
    __builtin_prefetch(k0p + (size_t)128 * H_DIM, 0, 1);  // next key group
    __builtin_prefetch(k1p + (size_t)128 * H_DIM, 0, 1);
    for (int k0 = 0; k0 < H_DIM; k0 += 32) {
      v16h a = load_frag(qp + k0);
      s0 = wmma_f16(a, load_frag(k0p + k0), s0);
      s1 = wmma_f16(a, load_frag(k1p + k0), s1);
    }

    // ---- per-chunk row max -> LDS  (row = r + 8*half, col = idx)
#pragma unroll
    for (int r = 0; r < 8; ++r) {
      float v = fmaxf(s0[r], s1[r]);
#pragma unroll
      for (int m = 1; m < 16; m <<= 1) v = fmaxf(v, __shfl_xor(v, m, 16));
      if (idx == 0) smax[wave][r + 8 * half] = v;
    }
    __syncthreads();  // barrier1: smax visible; also fences last iter's PV

    // ---- block-wide new max, exponentiate, publish P^T + partial sums
    float alpha[8];
#pragma unroll
    for (int r = 0; r < 8; ++r) {
      const int row = r + 8 * half;
      const float cmax = fmaxf(fmaxf(smax[0][row], smax[1][row]),
                               fmaxf(smax[2][row], smax[3][row]));
      const float mnew = fmaxf(mrow[r], cmax);
      alpha[r] = __expf(mrow[r] - mnew);
      mrow[r]  = mnew;
      const float p0 = __expf(s0[r] - mnew);
      const float p1 = __expf(s1[r] - mnew);
      float ps = p0 + p1;
#pragma unroll
      for (int m = 1; m < 16; m <<= 1) ps += __shfl_xor(ps, m, 16);
      if (idx == 0) ssum[wave][row] = ps;
      Pt[wave * 32 + idx][row]      = (_Float16)p0;
      Pt[wave * 32 + 16 + idx][row] = (_Float16)p1;
#pragma unroll
      for (int t = 0; t < 8; ++t) acc[t][r] *= alpha[r];
    }
    __syncthreads();  // barrier2: Pt/ssum visible

    // ---- running denominator
#pragma unroll
    for (int r = 0; r < 8; ++r) {
      const int row = r + 8 * half;
      srow[r] = srow[r] * alpha[r] +
                (ssum[0][row] + ssum[1][row] + ssum[2][row] + ssum[3][row]);
    }

    // ---- y += P(16x128) * V(128 x my 128-wide H slice): 4 K-subchunks
#pragma unroll
    for (int sc = 0; sc < 4; ++sc) {
      v16h pa;
#pragma unroll
      for (int j = 0; j < 8; ++j) {
        pa[j]     = Pt[sc * 32 + half * 8 + j][idx];
        pa[8 + j] = Pt[sc * 32 + 16 + half * 8 + j][idx];
      }
#pragma unroll
      for (int t = 0; t < 8; ++t) {
        const _Float16* vp =
            Vt + ((size_t)bb * H_DIM + h0 + t * 16 + idx) * S_LEN + kc +
            sc * 32 + half * 8;
        acc[t] = wmma_f16(pa, load_frag(vp), acc[t]);
      }
    }
  }

  // ---- normalize (one rcp per row) and store y (f16)
  float inv[8];
#pragma unroll
  for (int r = 0; r < 8; ++r) inv[r] = 1.0f / srow[r];
#pragma unroll
  for (int t = 0; t < 8; ++t) {
    const int col = h0 + t * 16 + idx;
#pragma unroll
    for (int r = 0; r < 8; ++r)
      Yh[(size_t)(bb * S_LEN + q0 + r + 8 * half) * H_DIM + col] =
          (_Float16)(acc[t][r] * inv[r]);
  }
}

// ---------------------------------------------------------------------------
// Output projection: out(f32) = Y(f16)[R,H] * Wo[H,D] + bo, Wo as WoT[D][H].
// One wave computes a 64x16 tile (4 row-tiles share each B fragment).
// ---------------------------------------------------------------------------
__global__ __launch_bounds__(32) void outproj_kernel(
    const _Float16* __restrict__ Y, const _Float16* __restrict__ WoT,
    const float* __restrict__ bo, float* __restrict__ Out) {
  const int l = threadIdx.x;
  const int half = l >> 4, idx = l & 15;
  const int col0 = blockIdx.x * 16;
  const int row0 = blockIdx.y * 64;

  v8f acc[4];
#pragma unroll
  for (int rt = 0; rt < 4; ++rt) acc[rt] = (v8f){};

  const _Float16* ar = Y   + (size_t)(row0 + idx) * H_DIM + half * 8;
  const _Float16* br = WoT + (size_t)(col0 + idx) * H_DIM + half * 8;
  for (int k0 = 0; k0 < H_DIM; k0 += 32) {
    const v16h bf = load_frag(br + k0);
#pragma unroll
    for (int rt = 0; rt < 4; ++rt)
      acc[rt] = wmma_f16(load_frag(ar + (size_t)rt * 16 * H_DIM + k0), bf,
                         acc[rt]);
  }
  const float bvl = bo[col0 + idx];
#pragma unroll
  for (int rt = 0; rt < 4; ++rt) {
#pragma unroll
    for (int r = 0; r < 8; ++r)
      Out[(size_t)(row0 + rt * 16 + r + 8 * half) * D_DIM + col0 + idx] =
          acc[rt][r] + bvl;
  }
}

// ---------------------------------------------------------------------------
extern "C" void kernel_launch(void* const* d_in, const int* in_sizes, int n_in,
                              void* d_out, int out_size, void* d_ws,
                              size_t ws_size, hipStream_t stream) {
  const float* x  = (const float*)d_in[0];
  const float* Wq = (const float*)d_in[1];
  const float* bq = (const float*)d_in[2];
  const float* Wk = (const float*)d_in[3];
  const float* bk = (const float*)d_in[4];
  const float* Wv = (const float*)d_in[5];
  const float* bv = (const float*)d_in[6];
  const float* Wo = (const float*)d_in[7];
  const float* bo = (const float*)d_in[8];
  float* out = (float*)d_out;

  char* ws = (char*)d_ws;
  const size_t szW = (size_t)512 * 512 * sizeof(_Float16);             // 512 KB
  const size_t szT = (size_t)B_SZ * S_LEN * H_DIM * sizeof(_Float16);  // 16 MB
  _Float16* WqT = (_Float16*)(ws);
  _Float16* WkT = (_Float16*)(ws + szW);
  _Float16* WvT = (_Float16*)(ws + 2 * szW);
  _Float16* WoT = (_Float16*)(ws + 3 * szW);
  _Float16* Qh  = (_Float16*)(ws + 4 * szW);
  _Float16* Kh  = (_Float16*)(ws + 4 * szW + szT);
  _Float16* Vt  = (_Float16*)(ws + 4 * szW + 2 * szT);
  _Float16* Yh  = (_Float16*)(ws + 4 * szW + 3 * szT);

  hipLaunchKernelGGL(prep_weights_kernel, dim3(512 * 512 / 256), dim3(256), 0,
                     stream, Wq, Wk, Wv, Wo, WqT, WkT, WvT, WoT);

  const int R = B_SZ * S_LEN;
  dim3 pg(H_DIM / 16, R / 64);
  hipLaunchKernelGGL(proj_kernel, pg, dim3(32), 0, stream, x, WqT, bq, Qh, 0);
  hipLaunchKernelGGL(proj_kernel, pg, dim3(32), 0, stream, x, WkT, bk, Kh, 0);
  hipLaunchKernelGGL(proj_kernel, pg, dim3(32), 0, stream, x, WvT, bv, Vt, 1);

  hipLaunchKernelGGL(attn_kernel, dim3(S_LEN / 16, B_SZ), dim3(128), 0, stream,
                     Qh, Kh, Vt, Yh);

  hipLaunchKernelGGL(outproj_kernel, dim3(D_DIM / 16, R / 64), dim3(32), 0,
                     stream, Yh, WoT, bo, out);
}
